// LanguageModel_11742440587670
// MI455X (gfx1250) — compile-verified, matrix operand order checked by reading, not run on hardware
//
#include <hip/hip_runtime.h>

// ---------------------------------------------------------------------------
// LSTM language model for MI455X (gfx1250, wave32, WMMA bf16 -> f32)
// ---------------------------------------------------------------------------

#define BATCH   64
#define SEQ     512
#define EMBED   512
#define HIDDEN  1024
#define GATES   4096      // 4*HIDDEN
#define VOCAB   32000

typedef __attribute__((ext_vector_type(16))) __bf16 v16bf;
typedef __attribute__((ext_vector_type(8)))  float  v8f;
typedef __attribute__((ext_vector_type(4)))  int    v4i;
typedef unsigned short ushort_t;

#if defined(__HIP_DEVICE_COMPILE__)
typedef __attribute__((address_space(1))) v4i* gv4i_p;   // global int4*
typedef __attribute__((address_space(3))) v4i* lv4i_p;   // LDS int4*
#endif

// float -> bf16 (round to nearest even)
static __device__ __forceinline__ ushort_t f2bf(float f) {
    unsigned u = __float_as_uint(f);
    unsigned r = 0x7FFFu + ((u >> 16) & 1u);
    u += r;
    return (ushort_t)(u >> 16);
}

// Build a 16-element bf16 WMMA fragment from two contiguous 16B chunks
// (elements 0..7 from p, 8..15 from p+16) — matches the 16-bit A/B layout:
// lanes 0-15 hold K = kb..kb+7 (VGPR0-3) and kb+16..kb+23 (VGPR4-7).
static __device__ __forceinline__ v16bf load_frag(const ushort_t* p) {
    union { uint4 u[2]; v16bf v; } t;
    t.u[0] = *reinterpret_cast<const uint4*>(p);
    t.u[1] = *reinterpret_cast<const uint4*>(p + 16);
    return t.v;
}

static __device__ __forceinline__ float fast_sigmoid(float x) {
    return 1.0f / (1.0f + __expf(-x));
}
static __device__ __forceinline__ float fast_tanh(float x) {
    float e = __expf(2.0f * x);
    return (e - 1.0f) / (e + 1.0f);
}

// ---------------------------------------------------------------------------
// Weight convert + transpose: src f32 [K,N] -> dst bf16 [N,K]
// ---------------------------------------------------------------------------
__global__ void cvt_transpose_kernel(const float* __restrict__ src,
                                     ushort_t* __restrict__ dst,
                                     int K, int N) {
    long long i = (long long)blockIdx.x * blockDim.x + threadIdx.x;
    long long total = (long long)K * N;
    if (i >= total) return;
    int k = (int)(i / N);
    int n = (int)(i % N);
    dst[(size_t)n * K + k] = f2bf(src[i]);
}

// ---------------------------------------------------------------------------
// Embedding gather -> bf16: emb[(b*SEQ+s)*EMBED + e]
// ---------------------------------------------------------------------------
__global__ void embed_kernel(const int* __restrict__ tok,
                             const float* __restrict__ table,
                             ushort_t* __restrict__ emb) {
    long long i = (long long)blockIdx.x * blockDim.x + threadIdx.x;
    long long total = (long long)BATCH * SEQ * EMBED;
    if (i >= total) return;
    int row = (int)(i / EMBED);
    int e   = (int)(i % EMBED);
    int t   = tok[row];
    emb[i] = f2bf(table[(size_t)t * EMBED + e]);
}

// ---------------------------------------------------------------------------
// WMMA GEMM: C = A[M,K](bf16) * BT[N,K](bf16)^T + bias[N]  (f32 out)
// One wave computes a 16x64 strip (4 WMMA accumulators sharing one A frag).
// Software-pipelined: next k-step's fragments are fetched before the WMMAs
// consuming the current ones, keeping global_load_b128 in flight under WMMA.
// timeMajor!=0 remaps output row m=(b*SEQ+s) -> (s*BATCH+b) so the LSTM scan
// reads each timestep as one contiguous slab.
// Requires M%16==0, N%64==0, K%32==0.
// ---------------------------------------------------------------------------
__global__ void gemm_wmma_bias(const ushort_t* __restrict__ A,
                               const ushort_t* __restrict__ BT,
                               const float* __restrict__ bias,
                               float* __restrict__ C,
                               int M, int N, int K, int timeMajor) {
    int waveId = (int)((blockIdx.x * blockDim.x + threadIdx.x) >> 5);
    int lane   = threadIdx.x & 31;
    int tilesN = N >> 6;
    int total  = (M >> 4) * tilesN;
    if (waveId >= total) return;                 // uniform per wave

    int m0 = (waveId / tilesN) << 4;
    int n0 = (waveId % tilesN) << 6;

    v8f acc[4] = {};
    int nloc = lane & 15;
    int kb   = (lane >> 4) << 3;                 // 0 or 8
    const ushort_t* aBase = A + (size_t)(m0 + nloc) * K + kb;
    const ushort_t* bBase[4];
#pragma unroll
    for (int j = 0; j < 4; ++j)
        bBase[j] = BT + (size_t)(n0 + j * 16 + nloc) * K + kb;

    // prologue: fetch k=0 fragments
    v16bf a = load_frag(aBase);
    v16bf b[4];
#pragma unroll
    for (int j = 0; j < 4; ++j) b[j] = load_frag(bBase[j]);

    for (int k = 32; k < K; k += 32) {
        // issue next iteration's loads first (overlap with WMMA below)
        v16bf an = load_frag(aBase + k);
        v16bf bn[4];
#pragma unroll
        for (int j = 0; j < 4; ++j) bn[j] = load_frag(bBase[j] + k);
#pragma unroll
        for (int j = 0; j < 4; ++j)
            acc[j] = __builtin_amdgcn_wmma_f32_16x16x32_bf16(
                false, a, false, b[j], (short)0, acc[j], false, false);
        a = an;
#pragma unroll
        for (int j = 0; j < 4; ++j) b[j] = bn[j];
    }
#pragma unroll
    for (int j = 0; j < 4; ++j)
        acc[j] = __builtin_amdgcn_wmma_f32_16x16x32_bf16(
            false, a, false, b[j], (short)0, acc[j], false, false);

    int mbase = m0 + ((lane >> 4) << 3);
#pragma unroll
    for (int r = 0; r < 8; ++r) {
        int m = mbase + r;
        int orow = timeMajor ? (((m & (SEQ - 1)) * BATCH) + (m >> 9)) : m;
#pragma unroll
        for (int j = 0; j < 4; ++j) {
            int n = n0 + j * 16 + nloc;
            float v = acc[j][r];
            if (bias) v += bias[n];
            C[(size_t)orow * N + n] = v;
        }
    }
}

// ---------------------------------------------------------------------------
// Persistent LSTM scan.
// Grid: 64 WGs (one per 16-hidden-unit slice), 128 threads = 4 waves
// (one per 16-batch-row slice). Wh slice (64 gate-cols x 1024 K, bf16) is
// staged in LDS (128KB of the 320KB WGP LDS) — via async global->LDS when
// the toolchain exposes it. Per step each wave:
//   acc(i,f,g,o)[16x16] = x4[t] slice  (f32 init, time-major contiguous)
//   acc += h(bf16, global) @ Wh(bf16, LDS)      -- 32 K-steps of WMMA
//   c = sig(f)*c + sig(i)*tanh(g); h = sig(o)*tanh(c)
// then device-wide barrier (agent-scope atomics + s_sleep spin).
// ---------------------------------------------------------------------------
__global__ void lstm_scan_kernel(const float* __restrict__ x4,     // [SEQ, B, 4H]
                                 const ushort_t* __restrict__ WhT, // [4H, H] bf16
                                 ushort_t* __restrict__ hstate,    // [B, H] bf16
                                 ushort_t* __restrict__ hseq,      // [B*SEQ, H] bf16 or null
                                 unsigned* __restrict__ bar,
                                 int writeSeq) {
    extern __shared__ ushort_t lds[];            // [64][1024] bf16
    const int hs0  = blockIdx.x << 4;            // hidden-slice base
    const int tid  = threadIdx.x;
    const int lane = tid & 31;
    const int wave = tid >> 5;

    // ---- stage Wh slice into LDS: row r = g*16+n  <-  WhT row (g*H + hs0+n)
#if __has_builtin(__builtin_amdgcn_global_load_async_to_lds_b128)
    for (int cc = tid; cc < 64 * 128; cc += blockDim.x) {
        int r  = cc >> 7;
        int k8 = (cc & 127) << 3;
        int g  = r >> 4, n = r & 15;
        const ushort_t* gsrc = &WhT[((size_t)(g * HIDDEN + hs0 + n) << 10) + k8];
        ushort_t* ldst = &lds[(r << 10) + k8];
        __builtin_amdgcn_global_load_async_to_lds_b128(
            (gv4i_p)(void*)gsrc, (lv4i_p)(void*)ldst, 0, 0);
    }
#if __has_builtin(__builtin_amdgcn_s_wait_asynccnt)
    __builtin_amdgcn_s_wait_asynccnt(0);
#endif
#else
    for (int cc = tid; cc < 64 * 128; cc += blockDim.x) {
        int r  = cc >> 7;
        int k8 = (cc & 127) << 3;
        int g  = r >> 4, n = r & 15;
        *reinterpret_cast<uint4*>(&lds[(r << 10) + k8]) =
            *reinterpret_cast<const uint4*>(
                &WhT[((size_t)(g * HIDDEN + hs0 + n) << 10) + k8]);
    }
#endif
    __syncthreads();

    const int m0    = wave << 4;                 // batch-row base for this wave
    const int nloc  = lane & 15;
    const int kb    = (lane >> 4) << 3;
    const int mbase = m0 + ((lane >> 4) << 3);
    const ushort_t* aBase = hstate + (size_t)(m0 + nloc) * HIDDEN + kb;

    v8f c = {};                                  // cell state, persists across t

    for (int t = 0; t < SEQ; ++t) {
        v8f acc[4];
        // init accumulators from precomputed input projection x4[t] (time-major)
#pragma unroll
        for (int g = 0; g < 4; ++g) {
            int col = g * HIDDEN + hs0 + nloc;
#pragma unroll
            for (int r = 0; r < 8; ++r) {
                acc[g][r] = x4[((size_t)t * BATCH + mbase + r) * GATES + col];
            }
        }
        // prefetch next timestep's x4 slice (global_prefetch_b8)
        if (t + 1 < SEQ) {
#pragma unroll
            for (int g = 0; g < 4; ++g) {
                __builtin_prefetch(
                    &x4[((size_t)(t + 1) * BATCH + mbase) * GATES + g * HIDDEN + hs0 + nloc],
                    0, 1);
            }
        }
        // acc += h @ Wh  (A from global h, B from LDS) — pipelined A fetch
        v16bf a = load_frag(aBase);
        for (int k = 0; k < HIDDEN; k += 32) {
            v16bf an = a;
            if (k + 32 < HIDDEN) an = load_frag(aBase + k + 32);
#pragma unroll
            for (int g = 0; g < 4; ++g) {
                v16bf b = load_frag(&lds[((g * 16 + nloc) << 10) + kb + k]);
                acc[g] = __builtin_amdgcn_wmma_f32_16x16x32_bf16(
                    false, a, false, b, (short)0, acc[g], false, false);
            }
            a = an;
        }
        // element-wise LSTM cell + store h (bf16)
#pragma unroll
        for (int r = 0; r < 8; ++r) {
            float ig = fast_sigmoid(acc[0][r]);
            float fg = fast_sigmoid(acc[1][r]);
            float gg = fast_tanh(acc[2][r]);
            float og = fast_sigmoid(acc[3][r]);
            float cn = fg * c[r] + ig * gg;
            c[r] = cn;
            ushort_t hb = f2bf(og * fast_tanh(cn));
            int m = mbase + r;
            hstate[(size_t)m * HIDDEN + hs0 + nloc] = hb;
            if (writeSeq)
                hseq[((size_t)m * SEQ + t) * HIDDEN + hs0 + nloc] = hb;
        }
        // device-wide barrier: all 64 WGs arrive, then proceed
        __threadfence();
        __syncthreads();
        if (tid == 0) {
            __hip_atomic_fetch_add(bar, 1u, __ATOMIC_ACQ_REL, __HIP_MEMORY_SCOPE_AGENT);
            unsigned target = (unsigned)(t + 1) * gridDim.x;
            while (__hip_atomic_load(bar, __ATOMIC_ACQUIRE, __HIP_MEMORY_SCOPE_AGENT) < target) {
                __builtin_amdgcn_s_sleep(2);
            }
        }
        __syncthreads();
        __threadfence();
    }
}

// ---------------------------------------------------------------------------
// In-place row softmax: one block (256 thr) per row of [rows, N]
// ---------------------------------------------------------------------------
__global__ void softmax_kernel(float* __restrict__ logits, int N) {
    __shared__ float red[256];
    float* p = logits + (size_t)blockIdx.x * N;
    int tid = threadIdx.x;

    float m = -3.4e38f;
    for (int i = tid; i < N; i += 256) m = fmaxf(m, p[i]);
    red[tid] = m; __syncthreads();
    for (int s = 128; s > 0; s >>= 1) {
        if (tid < s) red[tid] = fmaxf(red[tid], red[tid + s]);
        __syncthreads();
    }
    m = red[0]; __syncthreads();

    float sum = 0.0f;
    for (int i = tid; i < N; i += 256) sum += __expf(p[i] - m);
    red[tid] = sum; __syncthreads();
    for (int s = 128; s > 0; s >>= 1) {
        if (tid < s) red[tid] += red[tid + s];
        __syncthreads();
    }
    float inv = 1.0f / red[0]; __syncthreads();

    for (int i = tid; i < N; i += 256) p[i] = __expf(p[i] - m) * inv;
}

// ---------------------------------------------------------------------------
// Host launch
// ---------------------------------------------------------------------------
static inline size_t align256(size_t x) { return (x + 255) & ~(size_t)255; }

extern "C" void kernel_launch(void* const* d_in, const int* in_sizes, int n_in,
                              void* d_out, int out_size, void* d_ws, size_t ws_size,
                              hipStream_t stream) {
    const int*   x     = (const int*)  d_in[0];
    const float* table = (const float*)d_in[1];
    const float* Wx0   = (const float*)d_in[2];
    const float* Wh0   = (const float*)d_in[3];
    const float* b0    = (const float*)d_in[4];
    const float* Wx1   = (const float*)d_in[5];
    const float* Wh1   = (const float*)d_in[6];
    const float* b1    = (const float*)d_in[7];
    const float* Wout  = (const float*)d_in[8];
    const float* bout  = (const float*)d_in[9];
    float* out = (float*)d_out;

    // ---- workspace layout ----
    char* ws = (char*)d_ws;
    size_t off = 0;
    unsigned* bar = (unsigned*)(ws + off);            off = align256(off + 256);
    ushort_t* WxT0 = (ushort_t*)(ws + off);           off = align256(off + (size_t)GATES * EMBED  * 2);
    ushort_t* WhT0 = (ushort_t*)(ws + off);           off = align256(off + (size_t)GATES * HIDDEN * 2);
    ushort_t* WxT1 = (ushort_t*)(ws + off);           off = align256(off + (size_t)GATES * HIDDEN * 2);
    ushort_t* WhT1 = (ushort_t*)(ws + off);           off = align256(off + (size_t)GATES * HIDDEN * 2);
    ushort_t* WoT  = (ushort_t*)(ws + off);           off = align256(off + (size_t)VOCAB * HIDDEN * 2);
    ushort_t* emb  = (ushort_t*)(ws + off);           off = align256(off + (size_t)BATCH * SEQ * EMBED * 2);
    ushort_t* h1sq = (ushort_t*)(ws + off);           off = align256(off + (size_t)BATCH * SEQ * HIDDEN * 2);
    ushort_t* hst  = (ushort_t*)(ws + off);           off = align256(off + (size_t)BATCH * HIDDEN * 2);
    float*    x4   = (float*)(ws + off);              off = align256(off + (size_t)BATCH * SEQ * GATES * 4);
    (void)ws_size; (void)in_sizes; (void)n_in; (void)out_size;

    const int THR = 256;
    const size_t ldsBytes = 64 * 1024 * sizeof(ushort_t);   // 128 KB Wh slice

    // 1) weight convert + transpose to bf16 [N,K]
    cvt_transpose_kernel<<<(unsigned)(((size_t)EMBED  * GATES + THR - 1) / THR), THR, 0, stream>>>(Wx0,  WxT0, EMBED,  GATES);
    cvt_transpose_kernel<<<(unsigned)(((size_t)HIDDEN * GATES + THR - 1) / THR), THR, 0, stream>>>(Wh0,  WhT0, HIDDEN, GATES);
    cvt_transpose_kernel<<<(unsigned)(((size_t)HIDDEN * GATES + THR - 1) / THR), THR, 0, stream>>>(Wx1,  WxT1, HIDDEN, GATES);
    cvt_transpose_kernel<<<(unsigned)(((size_t)HIDDEN * GATES + THR - 1) / THR), THR, 0, stream>>>(Wh1,  WhT1, HIDDEN, GATES);
    cvt_transpose_kernel<<<(unsigned)(((size_t)HIDDEN * VOCAB + THR - 1) / THR), THR, 0, stream>>>(Wout, WoT,  HIDDEN, VOCAB);

    // 2) embedding gather -> bf16
    embed_kernel<<<(unsigned)(((size_t)BATCH * SEQ * EMBED + THR - 1) / THR), THR, 0, stream>>>(x, table, emb);

    // 3) x4_0 = emb @ Wx0 + b0   (M=32768, K=512, N=4096) -> time-major [S,B,4H]
    {
        int M = BATCH * SEQ, N = GATES, K = EMBED;
        unsigned waves = (unsigned)((M >> 4) * (N >> 6));
        gemm_wmma_bias<<<(waves + 7) / 8, THR, 0, stream>>>(emb, WxT0, b0, x4, M, N, K, 1);
    }

    // 4) LSTM layer 0 scan (writes h1 sequence as bf16)
    (void)hipMemsetAsync(bar, 0, 256, stream);
    (void)hipMemsetAsync(hst, 0, (size_t)BATCH * HIDDEN * 2, stream);
    lstm_scan_kernel<<<HIDDEN / 16, 128, ldsBytes, stream>>>(x4, WhT0, hst, h1sq, bar, 1);

    // 5) x4_1 = h1 @ Wx1 + b1   (M=32768, K=1024, N=4096) -> time-major; x4 reused
    {
        int M = BATCH * SEQ, N = GATES, K = HIDDEN;
        unsigned waves = (unsigned)((M >> 4) * (N >> 6));
        gemm_wmma_bias<<<(waves + 7) / 8, THR, 0, stream>>>(h1sq, WxT1, b1, x4, M, N, K, 1);
    }

    // 6) LSTM layer 1 scan (only final hstate needed)
    (void)hipMemsetAsync(bar, 0, 256, stream);
    (void)hipMemsetAsync(hst, 0, (size_t)BATCH * HIDDEN * 2, stream);
    lstm_scan_kernel<<<HIDDEN / 16, 128, ldsBytes, stream>>>(x4, WhT1, hst, (ushort_t*)nullptr, bar, 0);

    // 7) logits = h_final @ Wout + bout   (M=64, K=1024, N=32000) -> d_out
    {
        int M = BATCH, N = VOCAB, K = HIDDEN;
        unsigned waves = (unsigned)((M >> 4) * (N >> 6));   // 2000 waves
        gemm_wmma_bias<<<(waves + 7) / 8, THR, 0, stream>>>(hst, WoT, bout, out, M, N, K, 0);
    }

    // 8) softmax rows in place
    softmax_kernel<<<BATCH, 256, 0, stream>>>(out, VOCAB);
}
